// SimpleDecoder_45724221833688
// MI455X (gfx1250) — matrix operand order repaired and checked by reading, not compiled, not run on hardware
//
#include <hip/hip_runtime.h>
#include <hip/hip_bf16.h>

// ---------------------------------------------------------------------------
// Problem constants (from reference): B=32, L=128, V=32000, H=512, E=1024
// G = 4H = 2048, F = H+E = 1536, M = B*L = 4096
// ---------------------------------------------------------------------------
#define BATCH 32
#define SEQL  128
#define VOCAB 32000
#define HID   512
#define EMB   1024
#define GATES 2048
#define FEAT  1536
#define MROWS 4096   // B*L

typedef __attribute__((ext_vector_type(16))) __bf16 v16bf;
typedef __attribute__((ext_vector_type(8)))  __bf16 v8bf;
typedef __attribute__((ext_vector_type(8)))  float  v8f;

// ---------------------------------------------------------------------------
// WMMA fragment loaders (wave32, 16x16x32 bf16 shapes, CDNA5 ISA layouts)
// A 16x32: lane<16 -> row=lane,   K = {0..7, 16..23}
//          lane>=16-> row=lane-16,K = {8..15,24..31}
// B 32x16: lane<16 -> col=lane,   K = {0..7, 16..23}
//          lane>=16-> col=lane-16,K = {8..15,24..31}
// C 16x16: lane<16 -> col=lane, rows 0..7 ; lane>=16 -> col=lane-16, rows 8..15
// ---------------------------------------------------------------------------
__device__ __forceinline__ v16bf load_a_frag(const __bf16* base, int ld) {
  const int lane = threadIdx.x & 31;
  const int row  = lane & 15;
  const int kofs = (lane & 16) ? 8 : 0;
  const __bf16* p = base + (size_t)row * ld + kofs;
  v8bf lo = *reinterpret_cast<const v8bf*>(p);        // K chunk +0
  v8bf hi = *reinterpret_cast<const v8bf*>(p + 16);   // K chunk +16
  v16bf a;
#pragma unroll
  for (int i = 0; i < 8; ++i) { a[i] = lo[i]; a[i + 8] = hi[i]; }
  return a;
}

__device__ __forceinline__ v16bf load_b_frag(const __bf16* base, int ld) {
  const int lane = threadIdx.x & 31;
  const int col  = lane & 15;
  const int kofs = (lane & 16) ? 8 : 0;
  const __bf16* p = base + (size_t)col * ld + kofs;
  v8bf lo = *reinterpret_cast<const v8bf*>(p);
  v8bf hi = *reinterpret_cast<const v8bf*>(p + 16);
  v16bf b;
#pragma unroll
  for (int i = 0; i < 8; ++i) { b[i] = lo[i]; b[i + 8] = hi[i]; }
  return b;
}

#define WMMA_BF16(acc, a, b) \
  __builtin_amdgcn_wmma_f32_16x16x32_bf16(false, (a), false, (b), (short)0, (acc), false, false)

__device__ __forceinline__ float fsigmoid(float x) { return 1.0f / (1.0f + __expf(-x)); }

// ---------------------------------------------------------------------------
// Prep kernels
// ---------------------------------------------------------------------------
__global__ void k_cvt_bf16(const float* __restrict__ src, __bf16* __restrict__ dst, int n) {
  int i = blockIdx.x * blockDim.x + threadIdx.x;
  if (i < n) dst[i] = (__bf16)src[i];
}

__global__ void k_bias_sum(const float* __restrict__ b1, const float* __restrict__ b2,
                           float* __restrict__ out, int n) {
  int i = blockIdx.x * blockDim.x + threadIdx.x;
  if (i < n) out[i] = b1[i] + b2[i];
}

// Build x = concat(emb_table[in_tok], sent) in bf16; row m = b*L + l
__global__ void k_build_x(const float* __restrict__ emb, const float* __restrict__ sent,
                          const int* __restrict__ tok, __bf16* __restrict__ X) {
  const int m = blockIdx.x;                 // 0..4095
  const int b = m >> 7, l = m & 127;
  const int tk = (l == 0) ? 2 : tok[b * SEQL + (l - 1)];   // BOS = 2
  const float* er = emb  + (size_t)tk * HID;
  const float* sr = sent + (size_t)b * EMB;
  __bf16* xr = X + (size_t)m * FEAT;
  for (int i = threadIdx.x; i < HID; i += blockDim.x) xr[i] = (__bf16)er[i];
  for (int i = threadIdx.x; i < EMB; i += blockDim.x) xr[HID + i] = (__bf16)sr[i];
}

// ---------------------------------------------------------------------------
// Software-pipelined 32x64 wave GEMM body:
//   C(32x64) += A(32xK) * B^T(64xK), A/B row-major bf16, acc f32.
// Loads for step k+32 are issued BEFORE the WMMAs consuming step k's
// fragments, so the per-WMMA s_wait covers only older loads (L2 latency
// hidden behind the matrix pipe).
// ---------------------------------------------------------------------------
template <int KDIM>
__device__ __forceinline__ void gemm_wave_32x64(const __bf16* __restrict__ A, int lda,
                                                const __bf16* __restrict__ Bm, int ldb,
                                                v8f acc[2][4]) {
  v16bf a[2], b[4];
#pragma unroll
  for (int mi = 0; mi < 2; ++mi) a[mi] = load_a_frag(A + (size_t)(mi * 16) * lda, lda);
#pragma unroll
  for (int j = 0; j < 4; ++j)    b[j]  = load_b_frag(Bm + (size_t)(j * 16) * ldb, ldb);

#pragma unroll
  for (int k = 0; k < KDIM; k += 32) {
    v16bf an[2], bn[4];
    if (k + 32 < KDIM) {                    // compile-time resolved (full unroll)
#pragma unroll
      for (int mi = 0; mi < 2; ++mi)
        an[mi] = load_a_frag(A + (size_t)(mi * 16) * lda + k + 32, lda);
#pragma unroll
      for (int j = 0; j < 4; ++j)
        bn[j] = load_b_frag(Bm + (size_t)(j * 16) * ldb + k + 32, ldb);
    }
#pragma unroll
    for (int mi = 0; mi < 2; ++mi)
#pragma unroll
      for (int j = 0; j < 4; ++j)
        acc[mi][j] = WMMA_BF16(acc[mi][j], a[mi], b[j]);
    if (k + 32 < KDIM) {
#pragma unroll
      for (int mi = 0; mi < 2; ++mi) a[mi] = an[mi];
#pragma unroll
      for (int j = 0; j < 4; ++j)    b[j]  = bn[j];
    }
  }
}

// ---------------------------------------------------------------------------
// Kernel 2: xg = x @ W_ih^T + (b_ih+b_hh), stored f32 as (L, B, G)
// block = 128 threads (4 waves); wave tile = 32 x 64; grid = (128, 8)
// ---------------------------------------------------------------------------
__global__ __launch_bounds__(128) void k_xgate_gemm(const __bf16* __restrict__ X,
                                                    const __bf16* __restrict__ Wih,
                                                    const float* __restrict__ bias,
                                                    float* __restrict__ XG) {
  const int wave = threadIdx.x >> 5;
  const int m0 = blockIdx.x * 32;
  const int n0 = blockIdx.y * 256 + wave * 64;
  v8f acc[2][4] = {};
  gemm_wave_32x64<FEAT>(X + (size_t)m0 * FEAT, FEAT,
                        Wih + (size_t)n0 * FEAT, FEAT, acc);

  const int lane = threadIdx.x & 31;
  const int cn = lane & 15;
  const int mb = (lane & 16) ? 8 : 0;
#pragma unroll
  for (int mi = 0; mi < 2; ++mi)
#pragma unroll
    for (int j = 0; j < 4; ++j) {
      const int n = n0 + 16 * j + cn;
      const float bn = bias[n];
#pragma unroll
      for (int r = 0; r < 8; ++r) {
        const int m = m0 + mi * 16 + mb + r;
        const int bb = m >> 7, ll = m & 127;          // m = b*L + l
        XG[(size_t)(ll * BATCH + bb) * GATES + n] = acc[mi][j][r] + bn;
      }
    }
}

// ---------------------------------------------------------------------------
// Kernel 3: persistent LSTM scan. 16 WGs (one per 32-hidden-unit chunk),
// 256 threads (8 waves). Per step per WG: WMMA M=32(batch) x N=128(gates) x K=512.
// c-state lives in LDS; h ping-pongs through global (L2). Cross-WG sync:
// global-atomic sense barrier + CDNA5 cluster split-barrier (NOP if unclustered).
// K loop is double-buffered: this is the serial critical path (128 steps).
// ---------------------------------------------------------------------------
__global__ __launch_bounds__(256) void k_lstm_scan(const __bf16* __restrict__ Whh,
                                                   const float* __restrict__ XG,
                                                   __bf16* __restrict__ Hping,   // 2 x B x H
                                                   __bf16* __restrict__ HS,      // (L*B) x H
                                                   unsigned* __restrict__ bar) { // [0]=cnt [1]=gen
  __shared__ float gates[BATCH * 128];   // 16 KB: this WG's slice of all 4 gate types
  __shared__ float cstate[BATCH * 32];   //  4 KB: persistent cell state
  const int wg = blockIdx.x;             // hidden chunk: units [32*wg, 32*wg+32)
  const int tid = threadIdx.x;
  const int wave = tid >> 5;
  const int nblk = gridDim.x;

  for (int i = tid; i < BATCH * 32; i += blockDim.x) cstate[i] = 0.0f;
  __syncthreads();

  const int mt = wave & 1;               // batch tile (rows 0..15 / 16..31)
  const int nb = (wave >> 1) * 32;       // 32-col group within the 128-wide slice
  const int lane = tid & 31;
  const int cn = lane & 15;
  const int mb = (lane & 16) ? 8 : 0;

  // This wave's two B panels (gate rows, constant over time steps)
  const __bf16* bpan[2];
#pragma unroll
  for (int j = 0; j < 2; ++j) {
    const int nn = nb + 16 * j;                          // 0..127 within slice
    const int gt = nn >> 5;                              // gate type i/f/g/o
    const int grow = gt * HID + wg * 32 + (nn & 31);     // global gate row
    bpan[j] = Whh + (size_t)grow * HID;
  }

  for (int t = 0; t < SEQL; ++t) {
    const __bf16* hprev = Hping + (size_t)(t & 1) * (BATCH * HID);
    __bf16* hnext = Hping + (size_t)((t + 1) & 1) * (BATCH * HID);
    const __bf16* apan = hprev + (size_t)(mt * 16) * HID;

    v8f acc[2] = {};
    v16bf a = load_a_frag(apan, HID);
    v16bf b0 = load_b_frag(bpan[0], HID);
    v16bf b1 = load_b_frag(bpan[1], HID);
#pragma unroll
    for (int k = 0; k < HID; k += 32) {
      v16bf an, bn0, bn1;
      if (k + 32 < HID) {
        an  = load_a_frag(apan + k + 32, HID);
        bn0 = load_b_frag(bpan[0] + k + 32, HID);
        bn1 = load_b_frag(bpan[1] + k + 32, HID);
      }
      acc[0] = WMMA_BF16(acc[0], a, b0);
      acc[1] = WMMA_BF16(acc[1], a, b1);
      if (k + 32 < HID) { a = an; b0 = bn0; b1 = bn1; }
    }
#pragma unroll
    for (int j = 0; j < 2; ++j)
#pragma unroll
      for (int r = 0; r < 8; ++r)
        gates[(mt * 16 + mb + r) * 128 + nb + 16 * j + cn] = acc[j][r];
    __syncthreads();

    // elementwise LSTM cell update for this WG's (batch, hidden) pairs
    for (int idx = tid; idx < BATCH * 32; idx += blockDim.x) {
      const int b = idx >> 5, j = idx & 31;
      const size_t xb = ((size_t)t * BATCH + b) * GATES + wg * 32 + j;
      float iv = gates[b * 128 + j]      + XG[xb];
      float fv = gates[b * 128 + 32 + j] + XG[xb + 512];
      float gv = gates[b * 128 + 64 + j] + XG[xb + 1024];
      float ov = gates[b * 128 + 96 + j] + XG[xb + 1536];
      iv = fsigmoid(iv); fv = fsigmoid(fv); ov = fsigmoid(ov); gv = tanhf(gv);
      const float c = fv * cstate[idx] + iv * gv;
      cstate[idx] = c;
      const __bf16 hb = (__bf16)(ov * tanhf(c));
      hnext[b * HID + wg * 32 + j] = hb;
      HS[((size_t)t * BATCH + b) * HID + wg * 32 + j] = hb;
    }
    __syncthreads();

    // grid-wide sense barrier (device scope) so every WG sees the full hnext
    if (tid == 0) {
      __threadfence();
      const unsigned gen = (unsigned)(t + 1);
      if (atomicAdd(&bar[0], 1u) == (unsigned)(nblk - 1)) {
        atomicExch(&bar[0], 0u);
        __threadfence();
        atomicExch(&bar[1], gen);
      } else {
        while (__hip_atomic_load(&bar[1], __ATOMIC_ACQUIRE, __HIP_MEMORY_SCOPE_AGENT) < gen)
          __builtin_amdgcn_s_sleep(2);
      }
    }
    // CDNA5 cluster split-barrier: one wave signals, all waves wait.
    // NOP when ClusterID==0 (un-clustered launch); correct protocol otherwise.
    if (wave == 0) asm volatile("s_barrier_signal -3" ::: "memory");
    asm volatile("s_barrier_wait -3" ::: "memory");
    __syncthreads();
  }
}

// ---------------------------------------------------------------------------
// Kernel 4: logits = hs @ W_out^T + b_out  -> out (B, L, V) f32
// block = 128 threads (4 waves); wave tile 32 x 64; grid = (128, 125)
// ---------------------------------------------------------------------------
__global__ __launch_bounds__(128) void k_logits_gemm(const __bf16* __restrict__ HS,
                                                     const __bf16* __restrict__ Wout,
                                                     const float* __restrict__ bout,
                                                     float* __restrict__ out) {
  const int wave = threadIdx.x >> 5;
  const int m0 = blockIdx.x * 32;
  const int n0 = blockIdx.y * 256 + wave * 64;
  v8f acc[2][4] = {};
  gemm_wave_32x64<HID>(HS + (size_t)m0 * HID, HID,
                       Wout + (size_t)n0 * HID, HID, acc);

  const int lane = threadIdx.x & 31;
  const int cn = lane & 15;
  const int mb = (lane & 16) ? 8 : 0;
#pragma unroll
  for (int mi = 0; mi < 2; ++mi)
#pragma unroll
    for (int j = 0; j < 4; ++j) {
      const int n = n0 + 16 * j + cn;
      const float bn = bout[n];
#pragma unroll
      for (int r = 0; r < 8; ++r) {
        const int m = m0 + mi * 16 + mb + r;    // m = t*B + b
        const int ll = m >> 5, bb = m & 31;
        out[((size_t)bb * SEQL + ll) * VOCAB + n] = acc[mi][j][r] + bn;
      }
    }
}

// ---------------------------------------------------------------------------
// Host-side launch
// ---------------------------------------------------------------------------
extern "C" void kernel_launch(void* const* d_in, const int* in_sizes, int n_in,
                              void* d_out, int out_size, void* d_ws, size_t ws_size,
                              hipStream_t stream) {
  (void)in_sizes; (void)n_in; (void)out_size; (void)ws_size;
  const float* sent  = (const float*)d_in[0];   // (B, E)
  const int*   tok   = (const int*)  d_in[1];   // (B, L)
  const float* emb   = (const float*)d_in[2];   // (V, H)
  const float* Wih   = (const float*)d_in[3];   // (G, F)
  const float* Whh   = (const float*)d_in[4];   // (G, H)
  const float* b_ih  = (const float*)d_in[5];
  const float* b_hh  = (const float*)d_in[6];
  const float* Wout  = (const float*)d_in[7];   // (V, H)
  const float* b_out = (const float*)d_in[8];
  float* out = (float*)d_out;

  // workspace carve-up (≈ 92 MB)
  char* ws = (char*)d_ws;
  size_t o = 0;
  auto carve = [&](size_t bytes) { char* p = ws + o; o += (bytes + 255) & ~(size_t)255; return p; };
  __bf16*  Wih_bf  = (__bf16*)carve((size_t)GATES * FEAT * 2);
  __bf16*  Whh_bf  = (__bf16*)carve((size_t)GATES * HID * 2);
  __bf16*  Wout_bf = (__bf16*)carve((size_t)VOCAB * HID * 2);
  __bf16*  X_bf    = (__bf16*)carve((size_t)MROWS * FEAT * 2);
  float*   XG      = (float*) carve((size_t)MROWS * GATES * 4);
  __bf16*  HS      = (__bf16*)carve((size_t)MROWS * HID * 2);
  __bf16*  Hping   = (__bf16*)carve((size_t)2 * BATCH * HID * 2);
  float*   bias    = (float*) carve((size_t)GATES * 4);
  unsigned* bar    = (unsigned*)carve(256);

  // --- prep: bf16 weight copies, bias sum, input assembly, state init ---
  {
    int n1 = GATES * FEAT, n2 = GATES * HID, n3 = VOCAB * HID;
    k_cvt_bf16<<<(n1 + 255) / 256, 256, 0, stream>>>(Wih,  Wih_bf,  n1);
    k_cvt_bf16<<<(n2 + 255) / 256, 256, 0, stream>>>(Whh,  Whh_bf,  n2);
    k_cvt_bf16<<<(n3 + 255) / 256, 256, 0, stream>>>(Wout, Wout_bf, n3);
    k_bias_sum<<<(GATES + 255) / 256, 256, 0, stream>>>(b_ih, b_hh, bias, GATES);
    k_build_x<<<MROWS, 256, 0, stream>>>(emb, sent, tok, X_bf);
    hipMemsetAsync(Hping, 0, (size_t)2 * BATCH * HID * 2, stream);  // h0 = 0
    hipMemsetAsync(bar, 0, 256, stream);                            // barrier state
  }

  // --- xg = x @ W_ih^T + bias  (WMMA bf16, f32 out) ---
  k_xgate_gemm<<<dim3(MROWS / 32, GATES / 256), 128, 0, stream>>>(X_bf, Wih_bf, bias, XG);

  // --- sequential LSTM scan: 16 persistent WGs, grid barrier each step ---
  k_lstm_scan<<<16, 256, 0, stream>>>(Whh_bf, XG, Hping, HS, bar);

  // --- logits = hs @ W_out^T + b_out (dominant GEMM, 134 GFLOP) ---
  k_logits_gemm<<<dim3(MROWS / 32, VOCAB / 256), 128, 0, stream>>>(HS, Wout_bf, b_out, out);
}
